// PerformancePredictor_49520972923281
// MI455X (gfx1250) — matrix operand order repaired and checked by reading, not compiled, not run on hardware
//
#include <hip/hip_runtime.h>
#include <hip/hip_bf16.h>

// ---------------------------------------------------------------------------
// GCN forward for MI455X (gfx1250, wave32).
// GEMMs use v_wmma_f32_16x16x32_bf16 with weights staged in LDS as packed
// bf16 pairs; scatter/aggregation uses float4 gathers + global f32 atomics.
// N (=20000) is a multiple of 16, so every active 16-row WMMA tile is fully
// in-bounds: no per-row guards, EXEC stays all-ones around the WMMAs.
// ---------------------------------------------------------------------------

typedef __attribute__((ext_vector_type(16))) __bf16        v16bf;
typedef __attribute__((ext_vector_type(8)))  float         v8f;
typedef __attribute__((ext_vector_type(8)))  unsigned int  v8u;

#define DEV __device__ __forceinline__

DEV unsigned short f2bf(float f) {
  unsigned u = __float_as_uint(f);
  unsigned r = u + 0x7fffu + ((u >> 16) & 1u);   // round-to-nearest-even
  return (unsigned short)(r >> 16);
}

static inline int cdiv(int a, int b) { return (a + b - 1) / b; }

// ------------------------------ small utility kernels ----------------------

__global__ void k_fill(float* __restrict__ p, float v, int n) {
  int i = blockIdx.x * blockDim.x + threadIdx.x;
  if (i < n) p[i] = v;
}

__global__ void k_deg_count(const int* __restrict__ dst, float* __restrict__ deg, int E) {
  int e = blockIdx.x * blockDim.x + threadIdx.x;
  if (e < E) atomicAdd(&deg[dst[e]], 1.0f);
}

__global__ void k_rsqrt_inplace(float* __restrict__ p, int n) {
  int i = blockIdx.x * blockDim.x + threadIdx.x;
  if (i < n) p[i] = rsqrtf(p[i]);
}

__global__ void k_f32_to_bf16(const float* __restrict__ in, unsigned short* __restrict__ out, int n) {
  int i = blockIdx.x * blockDim.x + threadIdx.x;
  if (i < n) out[i] = f2bf(in[i]);
}

// ------------------------------ WMMA GEMM ----------------------------------
// Hout[N, NOUT] = Abf[N, K] (bf16) x W[K, NOUT] (f32->bf16)
// Block: 256 threads = 8 waves; each wave owns one 16-row tile (128 rows/blk).
// grid.y selects a 128-column chunk whose packed-bf16 weights live in LDS.
// Requires N % 16 == 0.
template <int K, int NOUT>
__global__ __launch_bounds__(256) void k_gemm_wmma(
    const unsigned short* __restrict__ Abf, const float* __restrict__ W,
    float* __restrict__ Hout, int N) {
  constexpr int KPAD  = (K + 31) & ~31;
  constexpr int NKT   = KPAD / 32;                 // k-steps of 32
  constexpr int KP    = KPAD / 2;                  // packed k-pairs
  constexpr int CHUNK = (NOUT < 128) ? NOUT : 128; // LDS column chunk

  __shared__ unsigned int ldsW[KP * CHUNK];        // <= 128*128*4 = 64 KB

  const int colBase = blockIdx.y * CHUNK;

  // Stage W chunk: ldsW[kp*CHUNK + c] = pack(bf16(W[2kp, col]), bf16(W[2kp+1, col]))
  for (int idx = threadIdx.x; idx < KP * CHUNK; idx += blockDim.x) {
    int kp = idx / CHUNK;
    int c  = idx - kp * CHUNK;
    int k0 = kp << 1;
    int col = colBase + c;
    float f0 = (k0     < K) ? W[(size_t)k0 * NOUT + col]       : 0.0f;
    float f1 = (k0 + 1 < K) ? W[(size_t)(k0 + 1) * NOUT + col] : 0.0f;
    ldsW[idx] = (unsigned)f2bf(f0) | ((unsigned)f2bf(f1) << 16);
  }
  __syncthreads();

  const int lane = threadIdx.x & 31;
  const int wave = threadIdx.x >> 5;
  const int tile = blockIdx.x * (blockDim.x >> 5) + wave;
  const int row0 = tile * 16;
  if (row0 >= N) return;                 // whole-wave exit: EXEC stays all-ones

  const int r    = lane & 15;
  const int half = lane >> 4;
  const int row  = row0 + r;             // always valid: N % 16 == 0

  // A fragments (16x32 bf16, ISA 7.12.2 layout). Each lane's data per k-step
  // is two contiguous 8-element bf16 runs: k0 = kt*32 + half*8 and k0 + 16.
  const unsigned short* arow = Abf + (size_t)row * K + half * 8;
  v8u afrag[NKT];
#pragma unroll
  for (int kt = 0; kt < NKT; ++kt) {
    v8u a;
    if constexpr ((K & 7) == 0) {
      // row*K and half*8 are both 8-elem aligned -> 16B-aligned b128 loads
      uint4 lo = *(const uint4*)(arow + kt * 32);
      uint4 hi = *(const uint4*)(arow + kt * 32 + 16);
      a[0] = lo.x; a[1] = lo.y; a[2] = lo.z; a[3] = lo.w;
      a[4] = hi.x; a[5] = hi.y; a[6] = hi.z; a[7] = hi.w;
    } else {
#pragma unroll
      for (int v = 0; v < 8; ++v) {
        int kb = (v < 4) ? (v * 2) : (16 + (v - 4) * 2);
        int k  = kt * 32 + half * 8 + kb;
        a[v] = (k < K) ? *(const unsigned*)(arow + kt * 32 + kb) : 0u;
      }
    }
    afrag[kt] = a;
  }

  constexpr int NTILES = CHUNK / 16;
#pragma unroll
  for (int nt = 0; nt < NTILES; ++nt) {
    const int cc = nt * 16 + r;          // column within chunk (B/D lane col)
    v8f acc = {0.f, 0.f, 0.f, 0.f, 0.f, 0.f, 0.f, 0.f};
#pragma unroll
    for (int kt = 0; kt < NKT; ++kt) {
      v8u bp;
#pragma unroll
      for (int v = 0; v < 8; ++v)        // B 32x16: VGPR v holds K = (v>>2)*16 + half*8 + 2*(v&3)
        bp[v] = ldsW[(kt * 16 + (v >> 2) * 8 + half * 4 + (v & 3)) * CHUNK + cc];
      acc = __builtin_amdgcn_wmma_f32_16x16x32_bf16(
          false, __builtin_bit_cast(v16bf, afrag[kt]),
          false, __builtin_bit_cast(v16bf, bp),
          (short)0, acc, false, false);
    }
    // D: VGPR v -> row (row0 + half*8 + v), col = colBase + nt*16 + r.
    // Constexpr v*NOUT offsets off one base pointer -> plain immediate stores.
    float* outp = Hout + (size_t)(row0 + half * 8) * NOUT + (colBase + nt * 16 + r);
#pragma unroll
    for (int v = 0; v < 8; ++v)
      outp[v * NOUT] = acc[v];
  }
}

// ------------------------------ graph aggregation --------------------------

// agg = dinv^2 * h (self loop) + bias   (initializes every element)
__global__ void k_agg_init(const float* __restrict__ h, const float* __restrict__ dinv,
                           const float* __restrict__ bias, float* __restrict__ agg,
                           int N, int C) {
  int idx = blockIdx.x * blockDim.x + threadIdx.x;
  if (idx >= N * C) return;
  int n = idx / C, c = idx - n * C;
  float di = dinv[n];
  agg[idx] = di * di * h[idx] + bias[c];
}

// agg[dst] += dinv[src]*dinv[dst] * h[src], 4 channels per thread
__global__ void k_scatter(const int* __restrict__ src, const int* __restrict__ dst,
                          const float* __restrict__ dinv, const float* __restrict__ h,
                          float* __restrict__ agg, int E, int C) {
  int gid = blockIdx.x * blockDim.x + threadIdx.x;
  int cg = C >> 2;
  if (gid >= E * cg) return;
  int e  = gid / cg;
  int c4 = (gid - e * cg) << 2;
  int s = src[e], d = dst[e];
  float w = dinv[s] * dinv[d];
  const float4 hv = *(const float4*)(h + (size_t)s * C + c4);
  float* ap = agg + (size_t)d * C + c4;
  atomicAdd(ap + 0, w * hv.x);
  atomicAdd(ap + 1, w * hv.y);
  atomicAdd(ap + 2, w * hv.z);
  atomicAdd(ap + 3, w * hv.w);
}

// ------------------------------ batch norm ---------------------------------

__global__ __launch_bounds__(256) void k_bn_stats(const float* __restrict__ a,
                                                  float* __restrict__ sum,
                                                  float* __restrict__ sumsq,
                                                  int N, int C) {
  __shared__ float ssum[256];
  __shared__ float ssq[256];
  for (int c = threadIdx.x; c < C; c += blockDim.x) { ssum[c] = 0.f; ssq[c] = 0.f; }
  __syncthreads();
  const int ROWS = 64;
  int r0 = blockIdx.x * ROWS;
  for (int i = threadIdx.x; i < ROWS * C; i += blockDim.x) {
    int r = r0 + i / C;
    int c = i % C;
    if (r < N) {
      float v = a[(size_t)r * C + c];
      atomicAdd(&ssum[c], v);
      atomicAdd(&ssq[c], v * v);
    }
  }
  __syncthreads();
  for (int c = threadIdx.x; c < C; c += blockDim.x) {
    atomicAdd(&sum[c], ssum[c]);
    atomicAdd(&sumsq[c], ssq[c]);
  }
}

// y = relu((a-mu)*rsqrt(var+eps)*g + be) [+ resid]; writes f32 and/or bf16
__global__ void k_bn_apply(const float* __restrict__ agg, const float* __restrict__ sum,
                           const float* __restrict__ sumsq, const float* __restrict__ g,
                           const float* __restrict__ be, const float* __restrict__ resid,
                           float* __restrict__ outf, unsigned short* __restrict__ outb,
                           int N, int C) {
  int idx = blockIdx.x * blockDim.x + threadIdx.x;
  if (idx >= N * C) return;
  int c = idx % C;
  float invN = 1.0f / (float)N;
  float mu  = sum[c] * invN;
  float var = sumsq[c] * invN - mu * mu;
  float y = (agg[idx] - mu) * rsqrtf(var + 1e-5f) * g[c] + be[c];
  y = fmaxf(y, 0.0f);
  if (resid) y += resid[idx];
  if (outf) outf[idx] = y;
  if (outb) outb[idx] = f2bf(y);
}

__global__ void k_relu(const float* __restrict__ in, float* __restrict__ out, int n) {
  int i = blockIdx.x * blockDim.x + threadIdx.x;
  if (i < n) out[i] = fmaxf(in[i], 0.0f);
}

// ------------------------------ pooling + predictor ------------------------

__global__ void k_pool_count(const int* __restrict__ batch, float* __restrict__ cnt, int N) {
  int i = blockIdx.x * blockDim.x + threadIdx.x;
  if (i < N) atomicAdd(&cnt[batch[i]], 1.0f);
}

__global__ void k_pool_sum(const float* __restrict__ x, const int* __restrict__ batch,
                           float* __restrict__ out, int N, int C) {
  int idx = blockIdx.x * blockDim.x + threadIdx.x;
  if (idx >= N * C) return;
  int n = idx / C, c = idx - n * C;
  atomicAdd(&out[batch[n] * C + c], x[idx]);
}

// single block: z=[gmean|hmean] (32x192) -> relu@P1 (64) -> relu@P2 (32) -> P3 (1)
__global__ __launch_bounds__(256) void k_predictor(
    const float* __restrict__ gsum, const float* __restrict__ hsum,
    const float* __restrict__ cnt,
    const float* __restrict__ P1w, const float* __restrict__ P1b,
    const float* __restrict__ P2w, const float* __restrict__ P2b,
    const float* __restrict__ P3w, const float* __restrict__ P3b,
    float* __restrict__ out) {
  __shared__ float z [32 * 192];
  __shared__ float z1[32 * 64];
  __shared__ float z2[32 * 32];
  int t = threadIdx.x, bd = blockDim.x;
  for (int i = t; i < 32 * 192; i += bd) {
    int g = i / 192, c = i - g * 192;
    float v = (c < 64) ? gsum[g * 64 + c] : hsum[g * 128 + (c - 64)];
    z[i] = v / cnt[g];
  }
  __syncthreads();
  for (int i = t; i < 32 * 64; i += bd) {
    int g = i / 64, c = i - g * 64;
    float acc = P1b[c];
    for (int k = 0; k < 192; ++k) acc += z[g * 192 + k] * P1w[k * 64 + c];
    z1[i] = fmaxf(acc, 0.0f);
  }
  __syncthreads();
  for (int i = t; i < 32 * 32; i += bd) {
    int g = i / 32, c = i - g * 32;
    float acc = P2b[c];
    for (int k = 0; k < 64; ++k) acc += z1[g * 64 + k] * P2w[k * 32 + c];
    z2[i] = fmaxf(acc, 0.0f);
  }
  __syncthreads();
  for (int g = t; g < 32; g += bd) {
    float acc = P3b[0];
    for (int k = 0; k < 32; ++k) acc += z2[g * 32 + k] * P3w[k];
    out[g] = acc;
  }
}

// ------------------------------ launch -------------------------------------

extern "C" void kernel_launch(void* const* d_in, const int* in_sizes, int n_in,
                              void* d_out, int out_size, void* d_ws, size_t ws_size,
                              hipStream_t stream) {
  (void)in_sizes; (void)n_in; (void)out_size; (void)ws_size;
  constexpr int N = 20000, E = 320000, F = 12, H = 256, G = 32;
  constexpr int T = 256;

  const float* x   = (const float*)d_in[0];
  const int*   ei  = (const int*)d_in[1];
  const int*   src = ei;
  const int*   dst = ei + E;
  const int*   bat = (const int*)d_in[2];
  const float* W1 = (const float*)d_in[3];  const float* b1 = (const float*)d_in[4];
  const float* W2 = (const float*)d_in[5];  const float* b2 = (const float*)d_in[6];
  const float* W3 = (const float*)d_in[7];  const float* b3 = (const float*)d_in[8];
  const float* W4 = (const float*)d_in[9];  const float* b4 = (const float*)d_in[10];
  const float* g1 = (const float*)d_in[11]; const float* be1 = (const float*)d_in[12];
  const float* g2 = (const float*)d_in[13]; const float* be2 = (const float*)d_in[14];
  const float* g3 = (const float*)d_in[15]; const float* be3 = (const float*)d_in[16];
  const float* P1w = (const float*)d_in[17]; const float* P1b = (const float*)d_in[18];
  const float* P2w = (const float*)d_in[19]; const float* P2b = (const float*)d_in[20];
  const float* P3w = (const float*)d_in[21]; const float* P3b = (const float*)d_in[22];
  float* out = (float*)d_out;

  char* wp = (char*)d_ws;
  auto carve = [&](size_t bytes) -> char* {
    char* p = wp;
    wp += (bytes + 255) & ~(size_t)255;
    return p;
  };
  float*          dinv = (float*)carve((size_t)N * 4);
  float*          h    = (float*)carve((size_t)N * H * 4);
  float*          agg  = (float*)carve((size_t)N * H * 4);
  float*          x1f  = (float*)carve((size_t)N * H * 4);
  float*          x3f  = (float*)carve((size_t)N * 128 * 4);
  float*          x4f  = (float*)carve((size_t)N * 64 * 4);
  unsigned short* xb   = (unsigned short*)carve((size_t)N * F * 2);
  unsigned short* x1b  = (unsigned short*)carve((size_t)N * H * 2);
  unsigned short* x2b  = (unsigned short*)carve((size_t)N * H * 2);
  unsigned short* x3b  = (unsigned short*)carve((size_t)N * 128 * 2);
  float*          ssum = (float*)carve(256 * 4);
  float*          ssq  = (float*)carve(256 * 4);
  float*          gsum = (float*)carve(G * 64 * 4);
  float*          hsum = (float*)carve(G * 128 * 4);
  float*          cnt  = (float*)carve(G * 4);

  const int rowBlocks = cdiv(N, 128);  // 8 waves x 16 rows per block

  // degree^{-1/2} with self loops (deg counted over dst, init 1 for self loop)
  k_fill<<<cdiv(N, T), T, 0, stream>>>(dinv, 1.0f, N);
  k_deg_count<<<cdiv(E, T), T, 0, stream>>>(dst, dinv, E);
  k_rsqrt_inplace<<<cdiv(N, T), T, 0, stream>>>(dinv, N);

  k_f32_to_bf16<<<cdiv(N * F, T), T, 0, stream>>>(x, xb, N * F);

  // ---- Layer 1: 12 -> 256 ----
  k_gemm_wmma<12, 256><<<dim3(rowBlocks, 2), T, 0, stream>>>(xb, W1, h, N);
  k_agg_init<<<cdiv(N * H, T), T, 0, stream>>>(h, dinv, b1, agg, N, H);
  k_scatter<<<cdiv(E * (H / 4), T), T, 0, stream>>>(src, dst, dinv, h, agg, E, H);
  k_fill<<<1, T, 0, stream>>>(ssum, 0.0f, 256);
  k_fill<<<1, T, 0, stream>>>(ssq, 0.0f, 256);
  k_bn_stats<<<cdiv(N, 64), T, 0, stream>>>(agg, ssum, ssq, N, H);
  k_bn_apply<<<cdiv(N * H, T), T, 0, stream>>>(agg, ssum, ssq, g1, be1, nullptr, x1f, x1b, N, H);

  // ---- Layer 2: 256 -> 256 (+ residual x1) ----
  k_gemm_wmma<256, 256><<<dim3(rowBlocks, 2), T, 0, stream>>>(x1b, W2, h, N);
  k_agg_init<<<cdiv(N * H, T), T, 0, stream>>>(h, dinv, b2, agg, N, H);
  k_scatter<<<cdiv(E * (H / 4), T), T, 0, stream>>>(src, dst, dinv, h, agg, E, H);
  k_fill<<<1, T, 0, stream>>>(ssum, 0.0f, 256);
  k_fill<<<1, T, 0, stream>>>(ssq, 0.0f, 256);
  k_bn_stats<<<cdiv(N, 64), T, 0, stream>>>(agg, ssum, ssq, N, H);
  k_bn_apply<<<cdiv(N * H, T), T, 0, stream>>>(agg, ssum, ssq, g2, be2, x1f, nullptr, x2b, N, H);

  // ---- Layer 3: 256 -> 128 ----
  k_gemm_wmma<256, 128><<<dim3(rowBlocks, 1), T, 0, stream>>>(x2b, W3, h, N);
  k_agg_init<<<cdiv(N * 128, T), T, 0, stream>>>(h, dinv, b3, agg, N, 128);
  k_scatter<<<cdiv(E * 32, T), T, 0, stream>>>(src, dst, dinv, h, agg, E, 128);
  k_fill<<<1, T, 0, stream>>>(ssum, 0.0f, 256);
  k_fill<<<1, T, 0, stream>>>(ssq, 0.0f, 256);
  k_bn_stats<<<cdiv(N, 64), T, 0, stream>>>(agg, ssum, ssq, N, 128);
  k_bn_apply<<<cdiv(N * 128, T), T, 0, stream>>>(agg, ssum, ssq, g3, be3, nullptr, x3f, x3b, N, 128);

  // ---- Layer 4: 128 -> 64 (no BN) ----
  k_gemm_wmma<128, 64><<<dim3(rowBlocks, 1), T, 0, stream>>>(x3b, W4, h, N);
  k_agg_init<<<cdiv(N * 64, T), T, 0, stream>>>(h, dinv, b4, agg, N, 64);
  k_scatter<<<cdiv(E * 16, T), T, 0, stream>>>(src, dst, dinv, h, agg, E, 64);
  k_relu<<<cdiv(N * 64, T), T, 0, stream>>>(agg, x4f, N * 64);

  // ---- Pooling + predictor ----
  k_fill<<<cdiv(G * 64, T), T, 0, stream>>>(gsum, 0.0f, G * 64);
  k_fill<<<cdiv(G * 128, T), T, 0, stream>>>(hsum, 0.0f, G * 128);
  k_fill<<<1, T, 0, stream>>>(cnt, 0.0f, G);
  k_pool_count<<<cdiv(N, T), T, 0, stream>>>(bat, cnt, N);
  k_pool_sum<<<cdiv(N * 64, T), T, 0, stream>>>(x4f, bat, gsum, N, 64);
  k_pool_sum<<<cdiv(N * 128, T), T, 0, stream>>>(x3f, bat, hsum, N, 128);
  k_predictor<<<1, T, 0, stream>>>(gsum, hsum, cnt, P1w, P1b, P2w, P2b, P3w, P3b, out);
}
